// StaircaseSeqAttention_11072425689258
// MI455X (gfx1250) — matrix operand order repaired and checked by reading, not compiled, not run on hardware
//
#include <hip/hip_runtime.h>

// Relative-position causal attention (Transformer-XL skew), flash-attention
// style, bf16 WMMA (v_wmma_f32_16x16x32_bf16) with f32 accumulation.
// B=16, M=2048, D=128, ATTN_LIM=2048.

#define MSEQ 2048
#define DH   128
#define LIM  2048
#define SCALE_F 0.08838834764831845f   // 1/sqrt(128)

typedef __attribute__((ext_vector_type(16))) __bf16 v16bf;
typedef __attribute__((ext_vector_type(8)))  float  v8f;

union V16 {
  v16bf v;
  unsigned short u[16];
  uint4 q[2];
};

__device__ __forceinline__ unsigned short f2bf(float x) {
  unsigned int u = __float_as_uint(x);
  u += 0x7FFFu + ((u >> 16) & 1u);   // round-to-nearest-even
  return (unsigned short)(u >> 16);
}

__device__ __forceinline__ v16bf lds_v16(const unsigned short* p) {
  V16 t;
  t.q[0] = *(const uint4*)(p);
  t.q[1] = *(const uint4*)(p + 8);
  return t.v;
}

__launch_bounds__(128)
__global__ void relattn_fwd(const float* __restrict__ Q, const float* __restrict__ K,
                            const float* __restrict__ V, const float* __restrict__ PE,
                            float* __restrict__ Out)
{
  // Shared staging (bf16 tiles + per-wave f32 score scratch). ~108 KB.
  __shared__ unsigned short Ksh[64 * 128];     // K tile, row-major [k][d]
  __shared__ unsigned short Vts[128 * 64];     // V tile, transposed [d][k]
  __shared__ unsigned short Ppe[128 * 128];    // PE window, transposed [col][d]
  __shared__ float          Sbuf[4][16 * 64];  // per-wave context scores
  __shared__ float          Tbuf[4][16 * 80];  // per-wave positional scores
  __shared__ unsigned short Pbuf[4][16 * 64];  // per-wave probabilities (bf16)

  const int tid  = threadIdx.x;
  const int lane = tid & 31;
  const int w    = tid >> 5;      // wave id 0..3 -> 16-row strip
  const int h    = lane >> 4;     // half-wave (K/N split in WMMA layouts)
  const int lm   = lane & 15;
  const int b    = blockIdx.y;
  const int i0   = blockIdx.x * 64;
  const int is   = i0 + w * 16;

  const float* Qb = Q + (size_t)b * MSEQ * DH;
  const float* Kb = K + (size_t)b * MSEQ * DH;
  const float* Vb = V + (size_t)b * MSEQ * DH;

  // ---- Q strip as WMMA A-operands (4 chunks of K=32), loaded once ----
  // A layout: lane m holds row m; halves carry K {0..15} / {16..31}.
  v16bf Qa[4];
  {
    const float* qr = Qb + (size_t)(is + lm) * DH;
#pragma unroll
    for (int dc = 0; dc < 4; ++dc) {
      const float* p = qr + dc * 32 + h * 16;
      float4 a0 = *(const float4*)(p + 0);
      float4 a1 = *(const float4*)(p + 4);
      float4 b0 = *(const float4*)(p + 8);
      float4 b1 = *(const float4*)(p + 12);
      V16 t;
      t.u[0]=f2bf(a0.x); t.u[1]=f2bf(a0.y); t.u[2]=f2bf(a0.z); t.u[3]=f2bf(a0.w);
      t.u[4]=f2bf(a1.x); t.u[5]=f2bf(a1.y); t.u[6]=f2bf(a1.z); t.u[7]=f2bf(a1.w);
      t.u[8]=f2bf(b0.x); t.u[9]=f2bf(b0.y); t.u[10]=f2bf(b0.z); t.u[11]=f2bf(b0.w);
      t.u[12]=f2bf(b1.x); t.u[13]=f2bf(b1.y); t.u[14]=f2bf(b1.z); t.u[15]=f2bf(b1.w);
      Qa[dc] = t.v;
    }
  }

  v8f O[8] = {};                       // 16 x 128 output accumulator (f32)
  float mrow = -__builtin_inff();      // running row max  (row = lane&15)
  float lrow = 0.f;                    // running row sum

  const int nkt = blockIdx.x + 1;      // causal: k-tiles 0..i0/64
  for (int kt = 0; kt < nkt; ++kt) {
    const int k0   = kt * 64;
    const int jmin = (MSEQ - 64) - i0 + k0;   // lowest PE column for the block

    __syncthreads();
    // ---- cooperative LDS fills (f32 -> bf16) ----
#pragma unroll 4
    for (int it = 0; it < 16; ++it) {         // K tile: 64x128
      int idx = tid + it * 128;
      int r = idx >> 5, c = (idx & 31) << 2;
      float4 f = *(const float4*)(Kb + (size_t)(k0 + r) * DH + c);
      unsigned short* d = Ksh + r * 128 + c;
      d[0]=f2bf(f.x); d[1]=f2bf(f.y); d[2]=f2bf(f.z); d[3]=f2bf(f.w);
    }
#pragma unroll 4
    for (int it = 0; it < 16; ++it) {         // V tile, transposed store
      int idx = tid + it * 128;
      int r = idx >> 5, c = (idx & 31) << 2;
      float4 f = *(const float4*)(Vb + (size_t)(k0 + r) * DH + c);
      Vts[(c + 0) * 64 + r] = f2bf(f.x);
      Vts[(c + 1) * 64 + r] = f2bf(f.y);
      Vts[(c + 2) * 64 + r] = f2bf(f.z);
      Vts[(c + 3) * 64 + r] = f2bf(f.w);
    }
#pragma unroll 4
    for (int it = 0; it < 32; ++it) {         // PE window, transposed + OOB pad
      int idx = tid + it * 128;
      int dd = idx >> 5, c = (idx & 31) << 2;
      int col = jmin + c;
      float x0, x1, x2, x3;
      if (col + 3 < LIM) {
        float4 f = *(const float4*)(PE + (size_t)dd * LIM + col);
        x0 = f.x; x1 = f.y; x2 = f.z; x3 = f.w;
      } else {
        x0 = (col + 0 < LIM) ? PE[(size_t)dd * LIM + col + 0] : 0.f;
        x1 = (col + 1 < LIM) ? PE[(size_t)dd * LIM + col + 1] : 0.f;
        x2 = (col + 2 < LIM) ? PE[(size_t)dd * LIM + col + 2] : 0.f;
        x3 = (col + 3 < LIM) ? PE[(size_t)dd * LIM + col + 3] : 0.f;
      }
      Ppe[(c + 0) * 128 + dd] = f2bf(x0);
      Ppe[(c + 1) * 128 + dd] = f2bf(x1);
      Ppe[(c + 2) * 128 + dd] = f2bf(x2);
      Ppe[(c + 3) * 128 + dd] = f2bf(x3);
    }
    __syncthreads();

    // ---- context scores: S = Q K^T (16 x 64 strip, 16 WMMA) ----
#pragma unroll
    for (int ct = 0; ct < 4; ++ct) {
      v8f acc = {};
#pragma unroll
      for (int dc = 0; dc < 4; ++dc) {
        v16bf Bk = lds_v16(Ksh + (ct * 16 + lm) * 128 + dc * 32 + h * 16);
        acc = __builtin_amdgcn_wmma_f32_16x16x32_bf16(false, Qa[dc], false, Bk,
                                                      (short)0, acc, false, false);
      }
#pragma unroll
      for (int r = 0; r < 8; ++r)
        Sbuf[w][(r + 8 * h) * 64 + ct * 16 + lm] = acc[r];
    }

    // ---- positional scores: T = Q * PE_window (16 x 80, 20 WMMA) ----
    const int pcb = 48 - 16 * w;   // wave's window offset inside the shared PE block
#pragma unroll
    for (int nt = 0; nt < 5; ++nt) {
      v8f acc = {};
#pragma unroll
      for (int dc = 0; dc < 4; ++dc) {
        v16bf Bp = lds_v16(Ppe + (pcb + nt * 16 + lm) * 128 + dc * 32 + h * 16);
        acc = __builtin_amdgcn_wmma_f32_16x16x32_bf16(false, Qa[dc], false, Bp,
                                                      (short)0, acc, false, false);
      }
#pragma unroll
      for (int r = 0; r < 8; ++r)
        Tbuf[w][(r + 8 * h) * 80 + nt * 16 + lm] = acc[r];
    }
    asm volatile("s_wait_dscnt 0x0" ::: "memory");  // strip stores -> cross-lane reads

    // ---- online softmax with skew gather + causal mask ----
    {
      const int row = lm;
      const int cb  = h * 32;          // lanes<16: cols 0..31, lanes>=16: 32..63
      float* Sw = Sbuf[w];
      float* Tw = Tbuf[w];
      float rmax = -__builtin_inff();
#pragma unroll 8
      for (int c = 0; c < 32; ++c) {
        int cc = cb + c;
        float s = Sw[row * 64 + cc] + Tw[row * 80 + (15 - row) + cc];  // skew diag
        s = ((k0 + cc) <= (is + row)) ? s * SCALE_F : -__builtin_inff();
        Sw[row * 64 + cc] = s;
        rmax = fmaxf(rmax, s);
      }
      rmax = fmaxf(rmax, __shfl_xor(rmax, 16, 32));
      float mnew = fmaxf(mrow, rmax);
      float corr = __expf(mrow - mnew);
      float ps = 0.f;
#pragma unroll 8
      for (int c = 0; c < 32; ++c) {
        int cc = cb + c;
        float p = __expf(Sw[row * 64 + cc] - mnew);
        ps += p;
        Pbuf[w][row * 64 + cc] = f2bf(p);
      }
      ps += __shfl_xor(ps, 16, 32);
      lrow = lrow * corr + ps;
      mrow = mnew;
      // rescale O accumulators: row m = r + 8*half lives at VGPR r
      float cf[8];
#pragma unroll
      for (int r = 0; r < 8; ++r) cf[r] = __shfl(corr, r + 8 * h, 32);
#pragma unroll
      for (int vt = 0; vt < 8; ++vt)
#pragma unroll
        for (int r = 0; r < 8; ++r) O[vt][r] *= cf[r];
    }
    asm volatile("s_wait_dscnt 0x0" ::: "memory");  // Pbuf stores -> A-operand reads

    // ---- O += P V (16 WMMA) ----
    v16bf Pa[2];
#pragma unroll
    for (int kc = 0; kc < 2; ++kc) {
      const unsigned short* pp = Pbuf[w] + lm * 64 + kc * 32 + h * 16;
      V16 t;
      t.q[0] = *(const uint4*)(pp);        // K {0..7}+16h
      t.q[1] = *(const uint4*)(pp + 8);    // K {8..15}+16h
      Pa[kc] = t.v;
    }
#pragma unroll
    for (int vt = 0; vt < 8; ++vt) {
#pragma unroll
      for (int kc = 0; kc < 2; ++kc) {
        v16bf Bv = lds_v16(Vts + (vt * 16 + lm) * 64 + kc * 32 + h * 16);
        O[vt] = __builtin_amdgcn_wmma_f32_16x16x32_bf16(false, Pa[kc], false, Bv,
                                                        (short)0, O[vt], false, false);
      }
    }
  }

  // ---- epilogue: normalize by row sums and store ----
  float linv = 1.0f / lrow;
  float lf[8];
#pragma unroll
  for (int r = 0; r < 8; ++r) lf[r] = __shfl(linv, r + 8 * h, 32);
  float* Ob = Out + ((size_t)b * MSEQ + is) * DH;
#pragma unroll
  for (int vt = 0; vt < 8; ++vt)
#pragma unroll
    for (int r = 0; r < 8; ++r)
      Ob[(size_t)(r + 8 * h) * DH + vt * 16 + lm] = O[vt][r] * lf[r];
}

extern "C" void kernel_launch(void* const* d_in, const int* in_sizes, int n_in,
                              void* d_out, int out_size, void* d_ws, size_t ws_size,
                              hipStream_t stream) {
  const float* q  = (const float*)d_in[0];   // query  [16,2048,128]
  const float* k  = (const float*)d_in[1];   // key    [16,2048,128]
  const float* v  = (const float*)d_in[2];   // value  [16,2048,128]
  const float* pe = (const float*)d_in[3];   // key_pe [1,128,2048]
  float* out = (float*)d_out;                // [16,2048,128]
  dim3 grid(MSEQ / 64, 16);
  relattn_fwd<<<grid, 128, 0, stream>>>(q, k, v, pe, out);
}
